// ChannelProcessing_12034498363951
// MI455X (gfx1250) — matrix operand order repaired and checked by reading, not compiled, not run on hardware
//
#include <hip/hip_runtime.h>
#include <hip/hip_bf16.h>
#include <math.h>

// MI455X / gfx1250, wave32. All GEMMs via v_wmma_f32_16x16x32_f16.
// Block = 256 threads (8 waves). Each block computes a 32 x 512 output slab;
// wave w owns output columns [64w, 64w+64) -> 4 col-tiles x 2 row-tiles
// = 8 WMMA accumulators. B fragments are reused across both row-tiles.

typedef __attribute__((ext_vector_type(16))) _Float16 v16h;
typedef __attribute__((ext_vector_type(8)))  _Float16 v8h;
typedef __attribute__((ext_vector_type(8)))  float    v8f;

#define B_  8
#define N_  4096
#define C_  512
#define M_  (B_*N_)   // 32768 rows for every GEMM

__device__ __forceinline__ v16h cat8(v8h lo, v8h hi) {
  return __builtin_shufflevector(lo, hi, 0,1,2,3,4,5,6,7,8,9,10,11,12,13,14,15);
}

// A-frag per lane (row = lane%16, h = lane/16): K = 16h+[0..15] (contiguous 32B).
// B-frag per lane (col = lane%16, h = lane/16): K = 16h+[0..15] (contiguous 32B).
__device__ __forceinline__ void gemm_mainloop32(
    const _Float16* __restrict__ A, const _Float16* __restrict__ Wt,
    long row0, int t, v8f accLo[4], v8f accHi[4])
{
  const int lane  = t & 31;
  const int wave  = t >> 5;
  const int r     = lane & 15;
  const int h     = lane >> 4;
  const int ncol0 = wave * 64;
  const _Float16* Arow0 = A + (row0 + r) * C_ + 16 * h;
  const _Float16* Arow1 = A + (row0 + 16 + r) * C_ + 16 * h;

  for (int k0 = 0; k0 < C_; k0 += 32) {
    v16h a0 = cat8(*(const v8h*)(Arow0 + k0), *(const v8h*)(Arow0 + k0 + 8));
    v16h a1 = cat8(*(const v8h*)(Arow1 + k0), *(const v8h*)(Arow1 + k0 + 8));
    if (k0 + 64 < C_) {
      __builtin_prefetch(Arow0 + k0 + 64, 0, 3);   // global_prefetch_b8
      __builtin_prefetch(Arow1 + k0 + 64, 0, 3);
    }
#pragma unroll
    for (int j = 0; j < 4; ++j) {
      const _Float16* wrow = Wt + (long)(ncol0 + j * 16 + r) * C_ + k0 + 16 * h;
      v16h bf = *(const v16h*)wrow;   // 32B aligned, contiguous K run
      accLo[j] = __builtin_amdgcn_wmma_f32_16x16x32_f16(
          false, a0, false, bf, (short)0, accLo[j], false, false);
      accHi[j] = __builtin_amdgcn_wmma_f32_16x16x32_f16(
          false, a1, false, bf, (short)0, accHi[j], false, false);
    }
  }
}

// --- GEMM 1: q = xh @ Wq^T  (raw f32 store) --------------------------------
__global__ __launch_bounds__(256) void k_gemm_q(
    const _Float16* __restrict__ A, const _Float16* __restrict__ Wt,
    float* __restrict__ Q)
{
  v8f accLo[4] = {}, accHi[4] = {};
  const long row0 = (long)blockIdx.x * 32;
  gemm_mainloop32(A, Wt, row0, threadIdx.x, accLo, accHi);
  const int lane = threadIdx.x & 31, wave = threadIdx.x >> 5;
  const int r = lane & 15, h = lane >> 4, ncol0 = wave * 64;
#pragma unroll
  for (int j = 0; j < 4; ++j) {
    const int col = ncol0 + j * 16 + r;
#pragma unroll
    for (int v = 0; v < 8; ++v) {
      Q[(row0 + v + 8 * h) * C_ + col]      = accLo[j][v];
      Q[(row0 + 16 + v + 8 * h) * C_ + col] = accHi[j][v];
    }
  }
}

// --- GEMM 2: h = gelu_exact(xh @ fc1^T + b1), stored f16 -------------------
__global__ __launch_bounds__(256) void k_gemm_fc1(
    const _Float16* __restrict__ A, const _Float16* __restrict__ Wt,
    const float* __restrict__ bias, _Float16* __restrict__ Hout)
{
  v8f accLo[4] = {}, accHi[4] = {};
  const long row0 = (long)blockIdx.x * 32;
  gemm_mainloop32(A, Wt, row0, threadIdx.x, accLo, accHi);
  const int lane = threadIdx.x & 31, wave = threadIdx.x >> 5;
  const int r = lane & 15, h = lane >> 4, ncol0 = wave * 64;
#pragma unroll
  for (int j = 0; j < 4; ++j) {
    const int col = ncol0 + j * 16 + r;
    const float bv = bias[col];
#pragma unroll
    for (int v = 0; v < 8; ++v) {
      float x0 = accLo[j][v] + bv;
      float x1 = accHi[j][v] + bv;
      float g0 = 0.5f * x0 * (1.0f + erff(x0 * 0.70710678118654752f));
      float g1 = 0.5f * x1 * (1.0f + erff(x1 * 0.70710678118654752f));
      Hout[(row0 + v + 8 * h) * C_ + col]      = (_Float16)g0;
      Hout[(row0 + 16 + v + 8 * h) * C_ + col] = (_Float16)g1;
    }
  }
}

// --- GEMM 3: m = LN(h @ fc2^T + b2); out = attn*m; also transposed copy ----
__global__ __launch_bounds__(256) void k_gemm_fc2_ln(
    const _Float16* __restrict__ A, const _Float16* __restrict__ Wt,
    const float* __restrict__ bias,
    const float* __restrict__ attn,   // [B, C]
    const float* __restrict__ lng, const float* __restrict__ lnb,
    float* __restrict__ out,          // [B, N, C]
    float* __restrict__ attn_v)       // [B, C, N]  (== [B, nh, d, N])
{
  __shared__ float sM[32 * C_];             // 64 KB of the 320 KB WGP LDS
  __shared__ float ps[32 * 8], pss[32 * 8];
  __shared__ float rmu[32], rrs[32];
  v8f accLo[4] = {}, accHi[4] = {};
  const long row0 = (long)blockIdx.x * 32;
  const int t = threadIdx.x;
  gemm_mainloop32(A, Wt, row0, t, accLo, accHi);

  const int lane = t & 31, wave = t >> 5;
  const int r = lane & 15, h = lane >> 4, ncol0 = wave * 64;
#pragma unroll
  for (int j = 0; j < 4; ++j) {
    const int col = ncol0 + j * 16 + r;
    const float bv = bias[col];
#pragma unroll
    for (int v = 0; v < 8; ++v) {
      sM[(v + 8 * h) * C_ + col]      = accLo[j][v] + bv;
      sM[(16 + v + 8 * h) * C_ + col] = accHi[j][v] + bv;
    }
  }
  __syncthreads();

  // LayerNorm over each 512-wide row: 8 threads per row, 64 cols each
  const int lr  = t & 31;    // row 0..31
  const int seg = t >> 5;    // 0..7
  float s = 0.f, ss = 0.f;
  for (int c = seg * 64; c < seg * 64 + 64; ++c) {
    float xv = sM[lr * C_ + c]; s += xv; ss += xv * xv;
  }
  ps[lr * 8 + seg] = s; pss[lr * 8 + seg] = ss;
  __syncthreads();
  if (t < 32) {
    float S = 0.f, SS = 0.f;
    for (int i = 0; i < 8; ++i) { S += ps[t * 8 + i]; SS += pss[t * 8 + i]; }
    float mu  = S * (1.0f / 512.0f);
    float var = SS * (1.0f / 512.0f) - mu * mu;
    rmu[t] = mu; rrs[t] = rsqrtf(var + 1e-5f);
  }
  __syncthreads();

  const long rowg = row0 + lr;
  const long b    = rowg >> 12;          // / N_
  const long n    = rowg & (N_ - 1);
  const float mu = rmu[lr], rs = rrs[lr];
  for (int c = seg * 64; c < seg * 64 + 64; ++c) {
    float m = (sM[lr * C_ + c] - mu) * rs * lng[c] + lnb[c];
    float o = attn[b * C_ + c] * m;
    out[rowg * C_ + c] = o;
    attn_v[((b * C_ + c) << 12) + n] = o;   // [B, C, N] transpose
  }
}

// --- column softmax stats: per (b,c) max and sum(exp) over N tokens --------
__global__ __launch_bounds__(256) void k_colstat(
    const float* __restrict__ V, float* __restrict__ Mo, float* __restrict__ So)
{
  // grid = B*8 blocks; each block: 64 columns of one batch, 4 threads/column
  const int b = blockIdx.x >> 3;
  const int c = ((blockIdx.x & 7) << 6) + (threadIdx.x & 63);
  const int g = threadIdx.x >> 6;
  const float* col = V + ((long)b * N_) * C_ + c;
  float m = -3.4e38f, s = 0.f;
  for (int n = g; n < N_; n += 4) {
    float x = col[(long)n * C_];
    if (x > m) { s = s * expf(m - x) + 1.0f; m = x; }
    else        { s += expf(x - m); }
  }
  __shared__ float sm[256], ssum[256];
  sm[threadIdx.x] = m; ssum[threadIdx.x] = s;
  __syncthreads();
  if (g == 0) {
    for (int i = 1; i < 4; ++i) {
      float m2 = sm[threadIdx.x + 64 * i], s2 = ssum[threadIdx.x + 64 * i];
      if (m2 > m) { s = s * expf(m - m2) + s2; m = m2; }
      else        { s += s2 * expf(m2 - m); }
    }
    Mo[b * C_ + c] = m; So[b * C_ + c] = s;
  }
}

// --- k_pool[b,h,n] = mean_d softmax_n(x)[b,n,h*64+d] -----------------------
__global__ __launch_bounds__(256) void k_kpool(
    const float* __restrict__ x, const float* __restrict__ Mx,
    const float* __restrict__ Sx, float* __restrict__ kp)
{
  const long id = (long)blockIdx.x * 256 + threadIdx.x;   // (b*8+h)*N + n
  const int  n  = (int)(id & (N_ - 1));
  const int  hh = (int)((id >> 12) & 7);
  const long b  = id >> 15;
  const float* xr = x  + (b * N_ + n) * C_ + hh * 64;
  const float* Mr = Mx + b * C_ + hh * 64;
  const float* Sr = Sx + b * C_ + hh * 64;
  float acc = 0.f;
  for (int d = 0; d < 64; ++d) acc += expf(xr[d] - Mr[d]) / Sr[d];
  kp[id] = acc * (1.0f / 64.0f);
}

// --- attn[b,c] = temp[h] * sigmoid( (Σn exp(q-Mq)*kpool) / Sq ) ------------
__global__ __launch_bounds__(256) void k_attn(
    const float* __restrict__ q, const float* __restrict__ Mq,
    const float* __restrict__ Sq, const float* __restrict__ kp,
    const float* __restrict__ temp, float* __restrict__ attn)
{
  const int b = blockIdx.x >> 3;
  const int c = ((blockIdx.x & 7) << 6) + (threadIdx.x & 63);
  const int g = threadIdx.x >> 6;
  const float M = Mq[b * C_ + c];
  const float S = Sq[b * C_ + c];
  const float* col = q  + ((long)b * N_) * C_ + c;
  const float* kpc = kp + ((long)b * 8 + (c >> 6)) * N_;
  float acc = 0.f;
  for (int n = g; n < N_; n += 4) acc += expf(col[(long)n * C_] - M) * kpc[n];
  __shared__ float sa[256];
  sa[threadIdx.x] = acc;
  __syncthreads();
  if (g == 0) {
    for (int i = 1; i < 4; ++i) acc += sa[threadIdx.x + 64 * i];
    float rr = acc / S;
    attn[b * C_ + c] = temp[c >> 6] / (1.0f + expf(-rr));
  }
}

// --- f32 -> f16 conversion --------------------------------------------------
__global__ __launch_bounds__(256) void k_cvt(
    const float* __restrict__ s, _Float16* __restrict__ d, long n)
{
  long i = (long)blockIdx.x * 256 + threadIdx.x;
  if (i < n) d[i] = (_Float16)s[i];
}

// ---------------------------------------------------------------------------
extern "C" void kernel_launch(void* const* d_in, const int* in_sizes, int n_in,
                              void* d_out, int out_size, void* d_ws, size_t ws_size,
                              hipStream_t stream)
{
  (void)in_sizes; (void)n_in; (void)out_size; (void)ws_size;
  const float* x    = (const float*)d_in[0];
  const float* Wq   = (const float*)d_in[1];
  const float* temp = (const float*)d_in[2];
  const float* f1w  = (const float*)d_in[3];
  const float* f1b  = (const float*)d_in[4];
  const float* f2w  = (const float*)d_in[5];
  const float* f2b  = (const float*)d_in[6];
  const float* lng  = (const float*)d_in[7];
  const float* lnb  = (const float*)d_in[8];
  float* out = (float*)d_out;

  char* ws = (char*)d_ws;
  size_t off = 0;
  auto alloc = [&](size_t bytes) -> char* {
    char* p = ws + off; off += (bytes + 255) & ~(size_t)255; return p;
  };
  _Float16* xh  = (_Float16*)alloc((size_t)M_ * C_ * 2);   // 32 MB
  _Float16* wqh = (_Float16*)alloc((size_t)C_ * C_ * 2);
  _Float16* f1h = (_Float16*)alloc((size_t)C_ * C_ * 2);
  _Float16* f2h = (_Float16*)alloc((size_t)C_ * C_ * 2);
  float*    q32 = (float*)   alloc((size_t)M_ * C_ * 4);   // 64 MB
  _Float16* h16 = (_Float16*)alloc((size_t)M_ * C_ * 2);   // 32 MB
  float*    Mx  = (float*)   alloc((size_t)B_ * C_ * 4);
  float*    Sx  = (float*)   alloc((size_t)B_ * C_ * 4);
  float*    Mq  = (float*)   alloc((size_t)B_ * C_ * 4);
  float*    Sq  = (float*)   alloc((size_t)B_ * C_ * 4);
  float*    kp  = (float*)   alloc((size_t)B_ * 8 * N_ * 4);
  float*    at  = (float*)   alloc((size_t)B_ * C_ * 4);

  const long nx = (long)M_ * C_;
  k_cvt<<<dim3((unsigned)((nx + 255) / 256)), 256, 0, stream>>>(x,   xh,  nx);
  k_cvt<<<dim3((C_ * C_ + 255) / 256),        256, 0, stream>>>(Wq,  wqh, (long)C_ * C_);
  k_cvt<<<dim3((C_ * C_ + 255) / 256),        256, 0, stream>>>(f1w, f1h, (long)C_ * C_);
  k_cvt<<<dim3((C_ * C_ + 255) / 256),        256, 0, stream>>>(f2w, f2h, (long)C_ * C_);

  // q projection (WMMA)
  k_gemm_q<<<M_ / 32, 256, 0, stream>>>(xh, wqh, q32);

  // attention scalars
  k_colstat<<<B_ * 8, 256, 0, stream>>>(x,   Mx, Sx);
  k_colstat<<<B_ * 8, 256, 0, stream>>>(q32, Mq, Sq);
  k_kpool<<<(B_ * 8 * N_) / 256, 256, 0, stream>>>(x, Mx, Sx, kp);
  k_attn<<<B_ * 8, 256, 0, stream>>>(q32, Mq, Sq, kp, temp, at);

  // MLP (WMMA) with fused GELU, then fused LayerNorm + attn scaling + outputs
  k_gemm_fc1<<<M_ / 32, 256, 0, stream>>>(xh, f1h, f1b, h16);
  k_gemm_fc2_ln<<<M_ / 32, 256, 0, stream>>>(h16, f2h, f2b, at, lng, lnb,
                                             out, out + (size_t)M_ * C_);
}
